// Tactile3DEncoder_88931592831264
// MI455X (gfx1250) — matrix operand order repaired and checked by reading, not compiled, not run on hardware
//
#include <hip/hip_runtime.h>

// ---------------------------------------------------------------------------
// PointNet++ (Tactile3DEncoder) for MI455X / gfx1250, wave32, WMMA f16.
// All MLP matmuls use v_wmma_f32_16x16x32_f16 with activations resident in
// LDS; weights are pre-transposed to fp16 [Cout, Kpad] so each lane's B
// operand is two 8-half loads (k = 8h+s and 16+8h+s) (matches CDNA5 B-matrix layout).
// ---------------------------------------------------------------------------

typedef __attribute__((ext_vector_type(16))) _Float16 v16h;
typedef __attribute__((ext_vector_type(8)))  _Float16 v8h;
typedef __attribute__((ext_vector_type(8)))  float    v8f;

// ------------------------- point cloud assembly ----------------------------
__global__ void k_build(const float* __restrict__ gp,  // [B,4096,3]
                        const float* __restrict__ t1,  // [B,512,6]
                        const float* __restrict__ t2,  // [B,512,6]
                        int B, float* __restrict__ xyz0,   // [B,5120,3]
                        float* __restrict__ feat0) {       // [B,5120,5]
  const int N = 5120;
  int t = blockIdx.x * blockDim.x + threadIdx.x;
  if (t >= B * N) return;
  int b = t / N, n = t % N;
  float x, y, z, f0 = 0.f, f1 = 0.f, f2 = 0.f, f3 = 0.f, f4 = 0.f;
  if (n < 4096) {
    const float* p = gp + ((size_t)b * 4096 + n) * 3;
    x = p[0]; y = p[1]; z = p[2];
  } else {
    const float* src = (n < 4608) ? t1 : t2;
    int nn = (n < 4608) ? (n - 4096) : (n - 4608);
    const float* p = src + ((size_t)b * 512 + nn) * 6;
    x = p[0]; y = p[1]; z = p[2];
    f0 = p[3]; f1 = p[4]; f2 = p[5]; f3 = 0.f; f4 = 1.f;
  }
  float* xo = xyz0 + (size_t)t * 3;  xo[0] = x; xo[1] = y; xo[2] = z;
  float* fo = feat0 + (size_t)t * 5; fo[0] = f0; fo[1] = f1; fo[2] = f2; fo[3] = f3; fo[4] = f4;
}

// ----------------------- weight convert/transpose --------------------------
// w: [cin, cout] f32 row-major -> wt: [cout, kpad] fp16, zero-padded K.
__global__ void k_wcvt(const float* __restrict__ w, int cin, int cout, int kpad,
                       _Float16* __restrict__ wt) {
  int t = blockIdx.x * blockDim.x + threadIdx.x;
  if (t >= cout * kpad) return;
  int n = t / kpad, k = t % kpad;
  wt[t] = (_Float16)((k < cin) ? w[(size_t)k * cout + n] : 0.f);
}

// ----------------------------- FPS -----------------------------------------
// One block per batch; dist[] lives in LDS; argmax matches jnp.argmax
// (first index on ties).
__global__ __launch_bounds__(512) void k_fps(const float* __restrict__ xyz,
                                             int N, int npoint,
                                             int* __restrict__ out_idx) {
  __shared__ float dist[5120];
  __shared__ float rv[512];
  __shared__ int   ri[512];
  __shared__ int   curi;
  int b = blockIdx.x;
  const float* x = xyz + (size_t)b * N * 3;
  for (int i = threadIdx.x; i < N; i += blockDim.x) dist[i] = 1e10f;
  if (threadIdx.x == 0) curi = 0;
  __syncthreads();
  for (int s = 0; s < npoint; ++s) {
    int c = curi;
    if (threadIdx.x == 0) out_idx[b * npoint + s] = c;
    float cx = x[c * 3], cy = x[c * 3 + 1], cz = x[c * 3 + 2];
    float best = -1.f; int bi = 0x7fffffff;
    for (int i = threadIdx.x; i < N; i += blockDim.x) {
      float dx = x[i * 3] - cx, dy = x[i * 3 + 1] - cy, dz = x[i * 3 + 2] - cz;
      float d = dx * dx + dy * dy + dz * dz;
      float nd = fminf(dist[i], d);
      dist[i] = nd;
      if (nd > best) { best = nd; bi = i; }
    }
    rv[threadIdx.x] = best; ri[threadIdx.x] = bi;
    __syncthreads();
    for (int off = blockDim.x >> 1; off > 0; off >>= 1) {
      if ((int)threadIdx.x < off) {
        float v2 = rv[threadIdx.x + off]; int i2 = ri[threadIdx.x + off];
        if (v2 > rv[threadIdx.x] ||
            (v2 == rv[threadIdx.x] && i2 < ri[threadIdx.x])) {
          rv[threadIdx.x] = v2; ri[threadIdx.x] = i2;
        }
      }
      __syncthreads();
    }
    if (threadIdx.x == 0) curi = ri[0];
    __syncthreads();
  }
}

__global__ void k_gather_centers(const float* __restrict__ xyz,
                                 const int* __restrict__ fidx,
                                 int N, int S, int count,
                                 float* __restrict__ nx) {
  int t = blockIdx.x * blockDim.x + threadIdx.x;
  if (t >= count) return;
  int b = t / S;
  int i = fidx[t];
  const float* p = xyz + ((size_t)b * N + i) * 3;
  float* o = nx + (size_t)t * 3;
  o[0] = p[0]; o[1] = p[1]; o[2] = p[2];
}

// --------------------------- ball query ------------------------------------
// One thread per (b,s): first-nsample-in-index-order within radius; pad with
// first hit (matches sort-then-truncate semantics of the reference).
__global__ void k_ball(const float* __restrict__ xyz, const float* __restrict__ cen,
                       int N, int S, int count, int ns, float r2,
                       int* __restrict__ idx) {
  int t = blockIdx.x * blockDim.x + threadIdx.x;
  if (t >= count) return;
  int b = t / S;
  const float* x = xyz + (size_t)b * N * 3;
  float cx = cen[(size_t)t * 3], cy = cen[(size_t)t * 3 + 1], cz = cen[(size_t)t * 3 + 2];
  int* o = idx + (size_t)t * ns;
  int cnt = 0, first = 0;
  for (int i = 0; i < N && cnt < ns; ++i) {
    float dx = x[i * 3] - cx, dy = x[i * 3 + 1] - cy, dz = x[i * 3 + 2] - cz;
    if (dx * dx + dy * dy + dz * dz < r2) {
      if (cnt == 0) first = i;
      o[cnt++] = i;
    }
  }
  for (; cnt < ns; ++cnt) o[cnt] = first;
}

// ------------------------ WMMA MLP layer core -------------------------------
// in : [M, Kpad] fp16 (LDS or global), row-major, Kpad % 32 == 0
// wt : [Cout, Kpad] fp16 (transposed weights)
// out (LAST=false): [M, Cout] fp16 LDS; (LAST=true): column-max -> outG[Cout]
template <bool LAST>
__device__ __forceinline__ void wmma_layer(
    const _Float16* __restrict__ in, int Kpad, int M,
    const _Float16* __restrict__ wt,
    const float* __restrict__ sc, const float* __restrict__ sh, int Cout,
    _Float16* __restrict__ outLds, float* __restrict__ outG) {
  const int lane  = threadIdx.x & 31;
  const int wave  = threadIdx.x >> 5;
  const int nwave = blockDim.x >> 5;
  const int lhalf = lane >> 4;   // K-half select (A) / K-half select (B)
  const int l16   = lane & 15;   // row (A) / column (B,D)

  for (int nt = wave; nt < (Cout >> 4); nt += nwave) {
    const int n = nt * 16 + l16;
    const float scv = sc[n], shv = sh[n];
    float cmax = 0.0f;
    for (int mt = 0; mt < (M >> 4); ++mt) {
      v8f acc = {};
      const int arow = mt * 16 + l16;
      for (int kt = 0; kt < (Kpad >> 5); ++kt) {
        const _Float16* ap = in + (size_t)arow * Kpad + kt * 32 + lhalf * 8;
        v8h a0 = *(const v8h*)ap;
        v8h a1 = *(const v8h*)(ap + 16);
        const _Float16* bp = wt + (size_t)n * Kpad + kt * 32 + lhalf * 8;
        v8h b0 = *(const v8h*)bp;
        v8h b1 = *(const v8h*)(bp + 16);
        v16h av, bv;
#pragma unroll
        for (int i = 0; i < 8; ++i) {
          av[i] = a0[i]; av[8 + i] = a1[i];
          bv[i] = b0[i]; bv[8 + i] = b1[i];
        }
        acc = __builtin_amdgcn_wmma_f32_16x16x32_f16(
            false, av, false, bv, (short)0, acc, false, false);
      }
      const int mbase = mt * 16 + lhalf * 8;
#pragma unroll
      for (int r = 0; r < 8; ++r) {
        float v = acc[r] * scv + shv;
        v = v > 0.f ? v : 0.f;
        if (LAST) {
          cmax = v > cmax ? v : cmax;
        } else {
          outLds[(size_t)(mbase + r) * Cout + n] = (_Float16)v;
        }
      }
    }
    if (LAST) {
      float o = __shfl_xor(cmax, 16, 32);  // combine rows m / m+8 halves
      cmax = cmax > o ? cmax : o;
      if (lane < 16) outG[n] = cmax;
    }
  }
}

// ---------------- fused gather + 3-layer MLP + maxpool (stages 0/1) --------
__global__ __launch_bounds__(128) void k_group_mlp(
    const float* __restrict__ xyz, int N,
    const float* __restrict__ centers,
    const float* __restrict__ feats, int Cf,
    const int* __restrict__ ballidx,
    int S, int ns, int Kpad0,
    const _Float16* __restrict__ wt0, const float* __restrict__ sc0, const float* __restrict__ sh0, int C1,
    const _Float16* __restrict__ wt1, const float* __restrict__ sc1, const float* __restrict__ sh1, int C2,
    const _Float16* __restrict__ wt2, const float* __restrict__ sc2, const float* __restrict__ sh2, int C3,
    float* __restrict__ out, int outStride, int outBase) {
  extern __shared__ _Float16 lds[];
  const int g = blockIdx.x;     // b*S + s
  const int b = g / S;
  _Float16* bufA = lds;
  const int szA = ns * (Kpad0 > C2 ? Kpad0 : C2);
  _Float16* bufB = lds + szA;

  const float cx = centers[(size_t)g * 3 + 0];
  const float cy = centers[(size_t)g * 3 + 1];
  const float cz = centers[(size_t)g * 3 + 2];
  const int* bi = ballidx + (size_t)g * ns;
  const int total = ns * Kpad0;
  for (int i = threadIdx.x; i < total; i += blockDim.x) {
    const int m = i / Kpad0, c = i % Kpad0;
    float v = 0.f;
    const int idx = bi[m];
    if (c < 3) {
      v = xyz[((size_t)b * N + idx) * 3 + c] - (c == 0 ? cx : (c == 1 ? cy : cz));
    } else if (c < 3 + Cf) {
      v = feats[((size_t)b * N + idx) * Cf + (c - 3)];
    }
    bufA[i] = (_Float16)v;
  }
  __syncthreads();
  wmma_layer<false>(bufA, Kpad0, ns, wt0, sc0, sh0, C1, bufB, nullptr);
  __syncthreads();
  wmma_layer<false>(bufB, C1, ns, wt1, sc1, sh1, C2, bufA, nullptr);
  __syncthreads();
  wmma_layer<true>(bufA, C2, ns, wt2, sc2, sh2, C3, nullptr,
                   out + (size_t)g * outStride + outBase);
}

// ----------------------------- stage-2 prep --------------------------------
__global__ void k_build_g2(const float* __restrict__ xyz2,  // [B,128,3]
                           const float* __restrict__ feats2,// [B,128,640]
                           int tot, _Float16* __restrict__ g2) { // [B,128,672]
  int t = blockIdx.x * blockDim.x + threadIdx.x;
  if (t >= tot) return;
  int c = t % 672, bm = t / 672;
  float v = 0.f;
  if (c < 3) v = xyz2[(size_t)bm * 3 + c];
  else if (c < 643) v = feats2[(size_t)bm * 640 + (c - 3)];
  g2[t] = (_Float16)v;
}

// --------------- stage-2 MLP (643->256->512->1024) + global maxpool --------
__global__ __launch_bounds__(128) void k_stage2(
    const _Float16* __restrict__ g2,
    const _Float16* __restrict__ wt0, const float* __restrict__ sc0, const float* __restrict__ sh0,
    const _Float16* __restrict__ wt1, const float* __restrict__ sc1, const float* __restrict__ sh1,
    const _Float16* __restrict__ wt2, const float* __restrict__ sc2, const float* __restrict__ sh2,
    float* __restrict__ out) {
  extern __shared__ _Float16 lds[];
  const int b = blockIdx.x;
  _Float16* buf1 = lds;              // 128*256 halves
  _Float16* buf2 = lds + 128 * 256;  // 128*512 halves
  const _Float16* in = g2 + (size_t)b * 128 * 672;
  wmma_layer<false>(in, 672, 128, wt0, sc0, sh0, 256, buf1, nullptr);
  __syncthreads();
  wmma_layer<false>(buf1, 256, 128, wt1, sc1, sh1, 512, buf2, nullptr);
  __syncthreads();
  wmma_layer<true>(buf2, 512, 128, wt2, sc2, sh2, 1024, nullptr,
                   out + (size_t)b * 1024);
}

// ---------------------------------------------------------------------------
extern "C" void kernel_launch(void* const* d_in, const int* in_sizes, int n_in,
                              void* d_out, int out_size, void* d_ws, size_t ws_size,
                              hipStream_t stream) {
  (void)in_sizes; (void)n_in; (void)out_size; (void)ws_size;
  const float* gp = (const float*)d_in[0];
  const float* t1 = (const float*)d_in[1];
  const float* t2 = (const float*)d_in[2];

  const int B = 32, N0 = 5120;

  // ---- workspace carve (~53 MB total) ----
  char* wsp = (char*)d_ws;
  auto carve = [&](size_t bytes) -> void* {
    void* p = (void*)wsp;
    wsp += (bytes + 255) & ~(size_t)255;
    return p;
  };
  float* xyz0   = (float*)carve((size_t)B * N0 * 3 * 4);
  float* feat0  = (float*)carve((size_t)B * N0 * 5 * 4);
  int*   fps1   = (int*)carve((size_t)B * 512 * 4);
  float* nx1    = (float*)carve((size_t)B * 512 * 3 * 4);
  float* feats1 = (float*)carve((size_t)B * 512 * 320 * 4);
  int*   fps2   = (int*)carve((size_t)B * 128 * 4);
  float* nx2    = (float*)carve((size_t)B * 128 * 3 * 4);
  float* feats2 = (float*)carve((size_t)B * 128 * 640 * 4);
  int*   ball   = (int*)carve((size_t)B * 512 * 128 * 4);
  _Float16* g2  = (_Float16*)carve((size_t)B * 128 * 672 * 2);

  // Layer dims: {cin, c1, c2, c3} per branch; param order = stage->branch->
  // layer->{scale,shift,w} (jax dict keys sorted), starting at d_in[3].
  static const int DIMS[7][4] = {
      {8, 32, 32, 64},   {8, 64, 64, 128},    {8, 64, 96, 128},
      {323, 64, 64, 128},{323, 128, 128, 256},{323, 128, 128, 256},
      {643, 256, 512, 1024}};

  const _Float16* wt[21]; const float* sc[21]; const float* sh[21];
  int kpadA[21], coutA[21];
  {
    int p = 3, li = 0;
    for (int br = 0; br < 7; ++br) {
      int cin = DIMS[br][0];
      for (int l = 0; l < 3; ++l, ++li, p += 3) {
        int cout = DIMS[br][l + 1];
        int kpad = ((cin + 31) / 32) * 32;
        sc[li] = (const float*)d_in[p + 0];
        sh[li] = (const float*)d_in[p + 1];
        const float* w = (const float*)d_in[p + 2];
        _Float16* dst = (_Float16*)carve((size_t)cout * kpad * 2);
        int tot = cout * kpad;
        k_wcvt<<<(tot + 255) / 256, 256, 0, stream>>>(w, cin, cout, kpad, dst);
        wt[li] = dst; kpadA[li] = kpad; coutA[li] = cout;
        cin = cout;
      }
    }
  }

  hipFuncSetAttribute(reinterpret_cast<const void*>(k_group_mlp),
                      hipFuncAttributeMaxDynamicSharedMemorySize, 160 * 1024);
  hipFuncSetAttribute(reinterpret_cast<const void*>(k_stage2),
                      hipFuncAttributeMaxDynamicSharedMemorySize, 208 * 1024);

  // ---- assemble point cloud ----
  k_build<<<(B * N0 + 255) / 256, 256, 0, stream>>>(gp, t1, t2, B, xyz0, feat0);

  // ---- stage 0: N=5120 -> S=512 ----
  k_fps<<<B, 512, 0, stream>>>(xyz0, N0, 512, fps1);
  k_gather_centers<<<(B * 512 + 255) / 256, 256, 0, stream>>>(
      xyz0, fps1, N0, 512, B * 512, nx1);
  {
    const float R[3] = {0.1f, 0.2f, 0.4f};
    const int   NS[3] = {16, 32, 128};
    int base = 0;
    for (int br = 0; br < 3; ++br) {
      int li = br * 3, ns = NS[br];
      k_ball<<<(B * 512 + 127) / 128, 128, 0, stream>>>(
          xyz0, nx1, N0, 512, B * 512, ns, R[br] * R[br], ball);
      int C1 = coutA[li], C2 = coutA[li + 1], C3 = coutA[li + 2], kp0 = kpadA[li];
      size_t lds = (size_t)(ns * ((kp0 > C2) ? kp0 : C2) + ns * C1) * 2;
      k_group_mlp<<<B * 512, 128, lds, stream>>>(
          xyz0, N0, nx1, feat0, 5, ball, 512, ns, kp0,
          wt[li], sc[li], sh[li], C1,
          wt[li + 1], sc[li + 1], sh[li + 1], C2,
          wt[li + 2], sc[li + 2], sh[li + 2], C3,
          feats1, 320, base);
      base += C3;
    }
  }

  // ---- stage 1: N=512 -> S=128 ----
  k_fps<<<B, 512, 0, stream>>>(nx1, 512, 128, fps2);
  k_gather_centers<<<(B * 128 + 255) / 256, 256, 0, stream>>>(
      nx1, fps2, 512, 128, B * 128, nx2);
  {
    const float R[3] = {0.2f, 0.4f, 0.8f};
    const int   NS[3] = {32, 64, 128};
    int base = 0;
    for (int br = 0; br < 3; ++br) {
      int li = 9 + br * 3, ns = NS[br];
      k_ball<<<(B * 128 + 127) / 128, 128, 0, stream>>>(
          nx1, nx2, 512, 128, B * 128, ns, R[br] * R[br], ball);
      int C1 = coutA[li], C2 = coutA[li + 1], C3 = coutA[li + 2], kp0 = kpadA[li];
      size_t lds = (size_t)(ns * ((kp0 > C2) ? kp0 : C2) + ns * C1) * 2;
      k_group_mlp<<<B * 128, 128, lds, stream>>>(
          nx1, 512, nx2, feats1, 320, ball, 128, ns, kp0,
          wt[li], sc[li], sh[li], C1,
          wt[li + 1], sc[li + 1], sh[li + 1], C2,
          wt[li + 2], sc[li + 2], sh[li + 2], C3,
          feats2, 640, base);
      base += C3;
    }
  }

  // ---- stage 2: global MLP + maxpool over 128 points -> d_out [32,1024] ----
  {
    int tot = B * 128 * 672;
    k_build_g2<<<(tot + 255) / 256, 256, 0, stream>>>(nx2, feats2, tot, g2);
    size_t lds = (size_t)(128 * 256 + 128 * 512) * 2;  // 192 KB
    k_stage2<<<B, 128, lds, stream>>>(
        g2, wt[18], sc[18], sh[18], wt[19], sc[19], sh[19],
        wt[20], sc[20], sh[20], (float*)d_out);
  }
}
